// AttnBlock3D_81252191306500
// MI455X (gfx1250) — matrix operand order repaired and checked by reading, not compiled, not run on hardware
//
#include <hip/hip_runtime.h>

// Problem constants (b=2, c=512, d=h=w=16 -> n=4096, 32 groups of 16 ch)
#define CC 512
#define NN 4096
#define BB 2
#define GG 32
#define CG 16

typedef _Float16 v16h __attribute__((ext_vector_type(16)));
typedef _Float16 v8h  __attribute__((ext_vector_type(8)));
typedef float    v8f  __attribute__((ext_vector_type(8)));
typedef _Float16 h4   __attribute__((ext_vector_type(4)));

// ---------------------------------------------------------------------------
// WMMA helpers. D(16x16 f32) = A(16x32 f16) * B(32x16 f16) + C.
// Layouts per CDNA5 ISA 7.12.2. All fragment loads are 2 x b128 per lane.
// ---------------------------------------------------------------------------
__device__ __forceinline__ v8f wmma16x16x32(v16h a, v16h b, v8f c) {
  return __builtin_amdgcn_wmma_f32_16x16x32_f16(false, a, false, b,
                                                (short)0, c, false, false);
}

__device__ __forceinline__ v16h cat8(v8h lo, v8h hi) {
  return __builtin_shufflevector(lo, hi, 0, 1, 2, 3, 4, 5, 6, 7,
                                 8, 9, 10, 11, 12, 13, 14, 15);
}

// A fragment (16x32) from row-major source, row stride ld (halfs).
// Lane L<16: row M=L, K {kh+0..7, kh+16..23} with kh = 8*(L>=16).
__device__ __forceinline__ v16h load_a_frag(const _Float16* src, int ld) {
  const int lane = threadIdx.x & 31;
  const _Float16* p = src + (size_t)(lane & 15) * ld + ((lane >> 4) << 3);
  return cat8(*(const v8h*)p, *(const v8h*)(p + 16));
}

// B fragment (32x16) from N-major source: element (k,n) at src[n*ldn + k].
// Lane L: col N=L%16, K {kh+0..7, kh+16..23} with kh = 8*(L>=16).
__device__ __forceinline__ v16h load_b_frag(const _Float16* src, int ldn) {
  const int lane = threadIdx.x & 31;
  const _Float16* p = src + (size_t)(lane & 15) * ldn + ((lane >> 4) << 3);
  return cat8(*(const v8h*)p, *(const v8h*)(p + 16));
}

// ---------------------------------------------------------------------------
// Kernel 0: weights f32 -> f16 (layout unchanged: [co][ci], used as A when
// co is the M dim and as N-major B when co is the N dim).
// ---------------------------------------------------------------------------
__global__ void __launch_bounds__(256) cvt_w_kernel(
    const float* __restrict__ wq, const float* __restrict__ wk,
    const float* __restrict__ wv, const float* __restrict__ wo,
    _Float16* __restrict__ wq16, _Float16* __restrict__ wk16,
    _Float16* __restrict__ wv16, _Float16* __restrict__ wo16) {
  const int idx = blockIdx.x * 256 + threadIdx.x;
  if (idx < CC * CC) {
    wq16[idx] = (_Float16)wq[idx];
    wk16[idx] = (_Float16)wk[idx];
    wv16[idx] = (_Float16)wv[idx];
    wo16[idx] = (_Float16)wo[idx];
  }
}

// ---------------------------------------------------------------------------
// Kernel 1: GroupNorm, output TRANSPOSED as f16 hnT[b][n][c] via LDS-tile
// transpose (group = exactly 16 channels, so each block owns 16 c-columns).
// ---------------------------------------------------------------------------
__global__ void __launch_bounds__(256) gn_kernel(
    const float* __restrict__ x, const float* __restrict__ gamma,
    const float* __restrict__ beta, _Float16* __restrict__ hnT) {
  const int g = blockIdx.x, b = blockIdx.y, t = threadIdx.x;
  const size_t base = ((size_t)b * CC + (size_t)g * CG) * NN;
  const float* xg = x + base;

  // ---- stats over 16 x 4096 = 65536 elements ----
  float s = 0.f, ss = 0.f;
#pragma unroll 4
  for (int it = 0; it < 64; ++it) {
    const float4 v = *(const float4*)(xg + it * 1024 + t * 4);
    s  += v.x + v.y + v.z + v.w;
    ss += v.x * v.x + v.y * v.y + v.z * v.z + v.w * v.w;
  }
  __shared__ float rs[256], rq[256];
  rs[t] = s; rq[t] = ss;
  __syncthreads();
  for (int k = 128; k > 0; k >>= 1) {
    if (t < k) { rs[t] += rs[t + k]; rq[t] += rq[t + k]; }
    __syncthreads();
  }
  __shared__ float sm, sr;
  if (t == 0) {
    const float mean = rs[0] * (1.f / 65536.f);
    const float var  = rq[0] * (1.f / 65536.f) - mean * mean;
    sm = mean; sr = rsqrtf(var + 1e-6f);
  }
  __syncthreads();
  const float mean = sm, rstd = sr;

  // ---- normalize + transpose: 16c x 64n LDS tiles ----
  const int cl = t >> 4, n4 = (t & 15) * 4;   // loader mapping
  const int nloc = t >> 2, cq = (t & 3) * 4;  // writer mapping
  float ga[4], be[4];
#pragma unroll
  for (int u = 0; u < 4; ++u) {
    ga[u] = gamma[g * CG + cq + u];
    be[u] = beta [g * CG + cq + u];
  }
  __shared__ float tile[16 * 68];             // pad 68 to dodge bank conflicts
  for (int nb = 0; nb < NN; nb += 64) {
    *(float4*)&tile[cl * 68 + n4] =
        *(const float4*)(xg + (size_t)cl * NN + nb + n4);
    __syncthreads();
    h4 o;
#pragma unroll
    for (int u = 0; u < 4; ++u)
      o[u] = (_Float16)((tile[(cq + u) * 68 + nloc] - mean) * rstd * ga[u] + be[u]);
    *(h4*)(hnT + ((size_t)b * NN + nb + nloc) * CC + g * CG + cq) = o;
    __syncthreads();
  }
}

// ---------------------------------------------------------------------------
// Kernel 2: q/k projections.  GEMM orientation M=i, N=co, K=ci:
//   A = hnT[i][ci] (row-major, contiguous), B = W[co][ci] (N-major, contiguous)
//   D[i][co] stored to qT/kT [b][n][c]; q pre-scaled by c^-0.5.
// ---------------------------------------------------------------------------
__global__ void __launch_bounds__(256) qk_kernel(
    const _Float16* __restrict__ wq16, const _Float16* __restrict__ wk16,
    const float* __restrict__ bq, const float* __restrict__ bk,
    const _Float16* __restrict__ hnT, _Float16* __restrict__ qT,
    _Float16* __restrict__ kT) {
  const int b = blockIdx.z >> 1, which = blockIdx.z & 1;  // 0 = q, 1 = k
  const _Float16* W   = which ? wk16 : wq16;
  const float*    bia = which ? bk : bq;
  _Float16*       dst = which ? kT : qT;
  const float     scl = which ? 1.f : 0.044194173824159216f;  // 512^-0.5

  const int wave = threadIdx.x >> 5;
  const int wm = wave & 3, wn = wave >> 2;
  const int ibase  = blockIdx.x * 128 + wm * 32;
  const int cobase = blockIdx.y * 128 + wn * 64;
  const _Float16* hb = hnT + (size_t)b * NN * CC;

  v8f acc[2][4] = {};
  for (int kk = 0; kk < 16; ++kk) {
    const int k0 = kk * 32;
    v16h a[2], bf[4];
#pragma unroll
    for (int ms = 0; ms < 2; ++ms)
      a[ms] = load_a_frag(hb + (size_t)(ibase + ms * 16) * CC + k0, CC);
#pragma unroll
    for (int ns = 0; ns < 4; ++ns)
      bf[ns] = load_b_frag(W + (size_t)(cobase + ns * 16) * CC + k0, CC);
#pragma unroll
    for (int ms = 0; ms < 2; ++ms)
#pragma unroll
      for (int ns = 0; ns < 4; ++ns)
        acc[ms][ns] = wmma16x16x32(a[ms], bf[ns], acc[ms][ns]);
  }

  const int lane = threadIdx.x & 31;
  const int nl = lane & 15, half = lane >> 4;
  _Float16* ob = dst + (size_t)b * NN * CC;
#pragma unroll
  for (int ns = 0; ns < 4; ++ns) {
    const int co = cobase + ns * 16 + nl;
    const float bias = bia[co];
#pragma unroll
    for (int ms = 0; ms < 2; ++ms)
#pragma unroll
      for (int r = 0; r < 8; ++r) {
        const int i = ibase + ms * 16 + (half << 3) + r;
        ob[(size_t)i * CC + co] = (_Float16)((acc[ms][ns][r] + bias) * scl);
      }
  }
}

// ---------------------------------------------------------------------------
// Kernel 3: v projection. Orientation M=co, N=i, K=ci:
//   A = Wv[co][ci] (row-major), B = hnT[i][ci] (N-major) -> D[co][i]
//   stored to v[b][c][n] (lanes along i: contiguous).
// ---------------------------------------------------------------------------
__global__ void __launch_bounds__(256) v_kernel(
    const _Float16* __restrict__ wv16, const float* __restrict__ bv,
    const _Float16* __restrict__ hnT, _Float16* __restrict__ vb) {
  const int b = blockIdx.z;
  const int wave = threadIdx.x >> 5;
  const int wm = wave & 3, wn = wave >> 2;
  const int mbase = blockIdx.y * 128 + wm * 32;   // co
  const int nbase = blockIdx.x * 128 + wn * 64;   // i
  const _Float16* hb = hnT + (size_t)b * NN * CC;

  v8f acc[2][4] = {};
  for (int kk = 0; kk < 16; ++kk) {
    const int k0 = kk * 32;
    v16h a[2], bf[4];
#pragma unroll
    for (int ms = 0; ms < 2; ++ms)
      a[ms] = load_a_frag(wv16 + (size_t)(mbase + ms * 16) * CC + k0, CC);
#pragma unroll
    for (int ns = 0; ns < 4; ++ns)
      bf[ns] = load_b_frag(hb + (size_t)(nbase + ns * 16) * CC + k0, CC);
#pragma unroll
    for (int ms = 0; ms < 2; ++ms)
#pragma unroll
      for (int ns = 0; ns < 4; ++ns)
        acc[ms][ns] = wmma16x16x32(a[ms], bf[ns], acc[ms][ns]);
  }

  const int lane = threadIdx.x & 31;
  const int nl = lane & 15, half = lane >> 4;
  _Float16* ob = vb + (size_t)b * CC * NN;
#pragma unroll
  for (int ms = 0; ms < 2; ++ms)
#pragma unroll
    for (int r = 0; r < 8; ++r) {
      const int co = mbase + ms * 16 + (half << 3) + r;
      const float bias = bv[co];
#pragma unroll
      for (int ns = 0; ns < 4; ++ns)
        ob[(size_t)co * NN + nbase + ns * 16 + nl] =
            (_Float16)(acc[ms][ns][r] + bias);
    }
}

// ---------------------------------------------------------------------------
// Kernel 4: single-pass flash attention. Workgroup = 32-row i-tile, 8 waves.
// Per 128-wide j-tile: S by WMMA (A from LDS q-tile, B = kT N-major),
// online-softmax (cross-wave max/sum via shfl + LDS), P -> LDS (f16),
// O-rescale, PV by WMMA into per-wave 64-channel VGPR accumulators.
// ---------------------------------------------------------------------------
__global__ void __launch_bounds__(256) attn_kernel(
    const _Float16* __restrict__ qT, const _Float16* __restrict__ kT,
    const _Float16* __restrict__ vb, _Float16* __restrict__ oT) {
  const int b = blockIdx.y;
  const int ibase = blockIdx.x * 32;
  const int tid = threadIdx.x;
  const int wave = tid >> 5, lane = tid & 31;
  const int nl = lane & 15, half = lane >> 4;

  __shared__ _Float16 qs[32 * 512];            // 32KB q tile (i-major)
  __shared__ _Float16 ps[32 * 128];            // 8KB P tile
  __shared__ float pmax[8][32], psum[8][32];
  __shared__ float mrow[32], lrow[32], srow[32];

  {  // stage q tile (contiguous 16384 halfs)
    const uint4* src = (const uint4*)(qT + ((size_t)b * NN + ibase) * CC);
    uint4* dst = (uint4*)qs;
#pragma unroll
    for (int it = 0; it < 8; ++it) dst[tid + it * 256] = src[tid + it * 256];
  }
  if (tid < 32) { mrow[tid] = -3.0e38f; lrow[tid] = 0.f; }
  __syncthreads();

  const _Float16* kf = kT + (size_t)b * NN * CC;
  const _Float16* vf = vb + (size_t)b * CC * NN;
  const int jcol  = wave * 16;  // wave's 16-col slice of the 128-wide S tile
  const int cbase = wave * 64;  // wave's private 64-channel slice of O

  v8f oacc[2][4] = {};
  for (int jt = 0; jt < 32; ++jt) {
    const int jbase = jt * 128;
    if (jt + 1 < 32)  // prefetch next k slice (global_prefetch_b8)
      __builtin_prefetch(kf + (size_t)(jbase + 128 + jcol) * CC + lane * 32, 0, 1);

    // ---- S = q^T k for this wave's 32x16 slice ----
    v8f s0 = {}, s1 = {};
    for (int kk = 0; kk < 16; ++kk) {
      const int k0 = kk * 32;
      v16h a0 = load_a_frag(qs + k0, 512);
      v16h a1 = load_a_frag(qs + 16 * 512 + k0, 512);
      v16h bf = load_b_frag(kf + (size_t)(jbase + jcol) * CC + k0, CC);
      s0 = wmma16x16x32(a0, bf, s0);
      s1 = wmma16x16x32(a1, bf, s1);
    }
    v8f sv[2] = {s0, s1};

    // ---- tile row-max partials (16-lane shfl reduce) ----
#pragma unroll
    for (int ms = 0; ms < 2; ++ms)
#pragma unroll
      for (int r = 0; r < 8; ++r) {
        float mx = sv[ms][r];
#pragma unroll
        for (int o = 1; o < 16; o <<= 1) mx = fmaxf(mx, __shfl_xor(mx, o, 32));
        if (nl == 0) pmax[wave][ms * 16 + (half << 3) + r] = mx;
      }
    __syncthreads();
    if (tid < 32) {
      const float mold = mrow[tid];
      float mloc = -3.0e38f;
#pragma unroll
      for (int w2 = 0; w2 < 8; ++w2) mloc = fmaxf(mloc, pmax[w2][tid]);
      const float mnew = fmaxf(mold, mloc);
      srow[tid] = __expf(mold - mnew);
      mrow[tid] = mnew;
    }
    __syncthreads();

    // ---- P = exp(S - m), partial row sums, O rescale ----
#pragma unroll
    for (int ms = 0; ms < 2; ++ms)
#pragma unroll
      for (int r = 0; r < 8; ++r) {
        const int row = ms * 16 + (half << 3) + r;
        const float p = __expf(sv[ms][r] - mrow[row]);
        ps[row * 128 + jcol + nl] = (_Float16)p;
        float e = p;
#pragma unroll
        for (int o = 1; o < 16; o <<= 1) e += __shfl_xor(e, o, 32);
        if (nl == 0) psum[wave][row] = e;
        const float sc = srow[row];
#pragma unroll
        for (int ns = 0; ns < 4; ++ns) oacc[ms][ns][r] *= sc;
      }
    __syncthreads();
    if (tid < 32) {
      float s = 0.f;
#pragma unroll
      for (int w2 = 0; w2 < 8; ++w2) s += psum[w2][tid];
      lrow[tid] = lrow[tid] * srow[tid] + s;
    }

    // ---- O += P * v^T  (wave-private 64-channel slice) ----
#pragma unroll
    for (int ks = 0; ks < 4; ++ks) {
      const int j0 = jbase + ks * 32;
      v16h a0 = load_a_frag(ps + ks * 32, 128);
      v16h a1 = load_a_frag(ps + 16 * 128 + ks * 32, 128);
      v16h bf[4];
#pragma unroll
      for (int ns = 0; ns < 4; ++ns)
        bf[ns] = load_b_frag(vf + (size_t)(cbase + ns * 16) * NN + j0, NN);
#pragma unroll
      for (int ns = 0; ns < 4; ++ns) {
        oacc[0][ns] = wmma16x16x32(a0, bf[ns], oacc[0][ns]);
        oacc[1][ns] = wmma16x16x32(a1, bf[ns], oacc[1][ns]);
      }
    }
    // next iteration's ps writes are separated from these reads by two
    // __syncthreads() (pmax/mrow phases), so no extra barrier needed here
  }
  __syncthreads();

  // ---- epilogue: O /= l, store oT[b][i][c] (coalesced along c) ----
  _Float16* ob = oT + ((size_t)b * NN + ibase) * CC;
#pragma unroll
  for (int ms = 0; ms < 2; ++ms)
#pragma unroll
    for (int r = 0; r < 8; ++r) {
      const int row = ms * 16 + (half << 3) + r;
      const float inv = 1.f / lrow[row];
#pragma unroll
      for (int ns = 0; ns < 4; ++ns)
        ob[(size_t)row * CC + cbase + ns * 16 + nl] =
            (_Float16)(oacc[ms][ns][r] * inv);
    }
}

// ---------------------------------------------------------------------------
// Kernel 5: output projection + residual.
// out[b,co,i] = x + bo[co] + sum_ci Wo[co,ci]*o[ci,i]; orientation M=i, N=co:
// A = oT[i][ci], B = Wo[co][ci] (N-major). Per lane the 8 D elements are
// consecutive i -> float4 ld/st of x/out.
// ---------------------------------------------------------------------------
__global__ void __launch_bounds__(256) proj_kernel(
    const _Float16* __restrict__ oT, const _Float16* __restrict__ wo16,
    const float* __restrict__ bo, const float* __restrict__ x,
    float* __restrict__ out) {
  const int b = blockIdx.z;
  const int wave = threadIdx.x >> 5;
  const int wm = wave & 3, wn = wave >> 2;
  const int ibase  = blockIdx.x * 128 + wm * 32;
  const int cobase = blockIdx.y * 128 + wn * 64;
  const _Float16* ob = oT + (size_t)b * NN * CC;

  v8f acc[2][4] = {};
  for (int kk = 0; kk < 16; ++kk) {
    const int k0 = kk * 32;
    v16h a[2], bf[4];
    a[0] = load_a_frag(ob + (size_t)ibase * CC + k0, CC);
    a[1] = load_a_frag(ob + (size_t)(ibase + 16) * CC + k0, CC);
#pragma unroll
    for (int ns = 0; ns < 4; ++ns)
      bf[ns] = load_b_frag(wo16 + (size_t)(cobase + ns * 16) * CC + k0, CC);
#pragma unroll
    for (int ms = 0; ms < 2; ++ms)
#pragma unroll
      for (int ns = 0; ns < 4; ++ns)
        acc[ms][ns] = wmma16x16x32(a[ms], bf[ns], acc[ms][ns]);
  }

  const int lane = threadIdx.x & 31;
  const int nl = lane & 15, half = lane >> 4;
#pragma unroll
  for (int ms = 0; ms < 2; ++ms)
#pragma unroll
    for (int ns = 0; ns < 4; ++ns) {
      const int co = cobase + ns * 16 + nl;
      const int i0 = ibase + ms * 16 + (half << 3);
      const float bias = bo[co];
      const float* xp = x   + ((size_t)b * CC + co) * NN + i0;
      float*       op = out + ((size_t)b * CC + co) * NN + i0;
      const float4 x0 = *(const float4*)(xp);
      const float4 x1 = *(const float4*)(xp + 4);
      float4 o0, o1;
      o0.x = x0.x + bias + acc[ms][ns][0];
      o0.y = x0.y + bias + acc[ms][ns][1];
      o0.z = x0.z + bias + acc[ms][ns][2];
      o0.w = x0.w + bias + acc[ms][ns][3];
      o1.x = x1.x + bias + acc[ms][ns][4];
      o1.y = x1.y + bias + acc[ms][ns][5];
      o1.z = x1.z + bias + acc[ms][ns][6];
      o1.w = x1.w + bias + acc[ms][ns][7];
      *(float4*)op       = o0;
      *(float4*)(op + 4) = o1;
    }
}

// ---------------------------------------------------------------------------
// Launch. Workspace (f16): hnT/qT/kT [n][c], v [c][n], oT [n][c]: 8 MiB each;
// f16 weights at +40 MiB (0.5 MiB each). Total 42 MiB.
// ---------------------------------------------------------------------------
extern "C" void kernel_launch(void* const* d_in, const int* in_sizes, int n_in,
                              void* d_out, int out_size, void* d_ws,
                              size_t ws_size, hipStream_t stream) {
  (void)in_sizes; (void)n_in; (void)out_size; (void)ws_size;
  const float* x     = (const float*)d_in[0];
  const float* gamma = (const float*)d_in[1];
  const float* beta  = (const float*)d_in[2];
  const float* wq    = (const float*)d_in[3];
  const float* bq    = (const float*)d_in[4];
  const float* wk    = (const float*)d_in[5];
  const float* bk    = (const float*)d_in[6];
  const float* wv    = (const float*)d_in[7];
  const float* bv    = (const float*)d_in[8];
  const float* wo    = (const float*)d_in[9];
  const float* bo    = (const float*)d_in[10];
  float* out = (float*)d_out;

  char* w = (char*)d_ws;
  const size_t MB = 1024 * 1024;
  _Float16* hnT  = (_Float16*)(w);
  _Float16* qT   = (_Float16*)(w + 8 * MB);
  _Float16* kT   = (_Float16*)(w + 16 * MB);
  _Float16* vbuf = (_Float16*)(w + 24 * MB);
  _Float16* oTb  = (_Float16*)(w + 32 * MB);
  _Float16* wq16 = (_Float16*)(w + 40 * MB);
  _Float16* wk16 = wq16 + CC * CC;
  _Float16* wv16 = wk16 + CC * CC;
  _Float16* wo16 = wv16 + CC * CC;

  cvt_w_kernel<<<dim3((CC * CC + 255) / 256), 256, 0, stream>>>(
      wq, wk, wv, wo, wq16, wk16, wv16, wo16);
  gn_kernel<<<dim3(GG, BB), 256, 0, stream>>>(x, gamma, beta, hnT);
  qk_kernel<<<dim3(NN / 128, CC / 128, BB * 2), 256, 0, stream>>>(
      wq16, wk16, bq, bk, hnT, qT, kT);
  v_kernel<<<dim3(NN / 128, CC / 128, BB), 256, 0, stream>>>(
      wv16, bv, hnT, vbuf);
  attn_kernel<<<dim3(NN / 32, BB), 256, 0, stream>>>(qT, kT, vbuf, oTb);
  proj_kernel<<<dim3(NN / 128, CC / 128, BB), 256, 0, stream>>>(
      oTb, wo16, bo, x, out);
}
